// GatedSelfAttention_50208167690537
// MI455X (gfx1250) — compile-verified, matrix operand order checked by reading, not run on hardware
//
#include <hip/hip_runtime.h>
#include <hip/hip_bf16.h>
#include <math.h>

#define HIDDEN   2560
#define N_HEADS  16
#define N_KV     4
#define HEAD_DIM 256
#define INNER    4096            // N_HEADS*HEAD_DIM
#define KVDIM    1024            // N_KV*HEAD_DIM
#define BATCH    2
#define SEQ      2048
#define TOKENS   (BATCH*SEQ)     // 4096

typedef __attribute__((ext_vector_type(16))) __bf16 v16bf;
typedef __attribute__((ext_vector_type(8)))  float  v8f;
typedef __attribute__((ext_vector_type(4)))  unsigned int u32x4;
typedef __attribute__((ext_vector_type(8)))  int    i32x8;
typedef __attribute__((ext_vector_type(4)))  int    i32x4;

#if __has_builtin(__builtin_amdgcn_tensor_load_to_lds)
#define HAVE_TDM 1
#else
#define HAVE_TDM 0
#endif

union FragU { v16bf v; uint4 u[2]; };
union Chunk { uint4 u; __bf16 h[8]; };

// Load a 16-bit WMMA A/B fragment from LDS for one lane.
// ISA layout (16-bit A 16x32): lanes 0-15 hold K={0..7,16..23}, lanes 16-31 K={8..15,24..31}
// => per-lane two contiguous 16-byte chunks at kbase and kbase+16.
__device__ __forceinline__ v16bf ldsFrag(const __bf16* rowPtr, int kbase) {
    FragU f;
    f.u[0] = *reinterpret_cast<const uint4*>(rowPtr + kbase);
    f.u[1] = *reinterpret_cast<const uint4*>(rowPtr + kbase + 16);
    return f.v;
}

__device__ __forceinline__ v8f wmma_bf16(v16bf a, v16bf b, v8f c) {
    return __builtin_amdgcn_wmma_f32_16x16x32_bf16(false, a, false, b, (short)0, c, false, false);
}

// ------------------------------------------------------------ TDM 2D tile load
// Builds D# group0/group1 per CDNA5 ISA 8.3/8.4 and issues TENSOR_LOAD_TO_LDS.
// 2D tile: tile_d1 rows x tile_d0 elements (bf16), row stride stride0 elements.
// This toolchain exposes the 6-arg builtin: (g0, g1, g2, g3, g4, cpol).
__device__ __forceinline__ void tdm_load_2d(unsigned int ldsAddr, const void* gptr,
                                            unsigned int tensor_d0, unsigned int tensor_d1,
                                            unsigned int tile_d0, unsigned int tile_d1,
                                            unsigned long long stride0) {
#if HAVE_TDM
    unsigned long long ga = (unsigned long long)(uintptr_t)gptr;
    u32x4 g0;
    g0[0] = 1u;                                              // count=1, user mode
    g0[1] = ldsAddr;                                         // LDS byte address
    g0[2] = (unsigned int)ga;                                // global_addr[31:0]
    g0[3] = (unsigned int)((ga >> 32) & 0x1FFFFFFu) | (2u << 30);  // addr[56:32], type=2
    i32x8 g1;
    g1[0] = 0x00010000;                                      // data_size=1 (2B), no mask
    g1[1] = (int)((tensor_d0 & 0xFFFFu) << 16);              // tensor_dim0[15:0]
    g1[2] = (int)((tensor_d0 >> 16) | ((tensor_d1 & 0xFFFFu) << 16));
    g1[3] = (int)((tensor_d1 >> 16) | (tile_d0 << 16));      // tile_dim0
    g1[4] = (int)(tile_d1 & 0xFFFFu);                        // tile_dim1 (tile_dim2=0)
    g1[5] = (int)(stride0 & 0xFFFFFFFFull);                  // tensor_dim0_stride[31:0]
    g1[6] = (int)((stride0 >> 32) & 0xFFFFull);              // stride[47:32], dim1_stride=0
    g1[7] = 0;
    i32x4 z4 = {0, 0, 0, 0};
    i32x8 z8 = {0, 0, 0, 0, 0, 0, 0, 0};
    __builtin_amdgcn_tensor_load_to_lds(g0, g1, z4, z4, z8, 0);
#else
    (void)ldsAddr; (void)gptr; (void)tensor_d0; (void)tensor_d1;
    (void)tile_d0; (void)tile_d1; (void)stride0;
#endif
}

__device__ __forceinline__ void tdm_wait() {
#if HAVE_TDM
    __builtin_amdgcn_s_wait_tensorcnt(0);
#endif
}

// ---------------------------------------------------------------- fp32 -> bf16
__global__ __launch_bounds__(256)
void cvt_f32_bf16(const float* __restrict__ in, __bf16* __restrict__ out, int n) {
    int i = (blockIdx.x * 256 + threadIdx.x) * 4;
    if (i + 3 < n) {
        float4 v = *reinterpret_cast<const float4*>(in + i);
        out[i + 0] = (__bf16)v.x;
        out[i + 1] = (__bf16)v.y;
        out[i + 2] = (__bf16)v.z;
        out[i + 3] = (__bf16)v.w;
    }
}

// ---------------------------------------------------------------- bf16 GEMM
// C[M,N] = A[M,K] * B[K,N]; 128x64 tile, 4 waves, each wave 32 rows x 64 cols
// => 8 WMMAs per 32-deep k-step per wave. A tile staged by TDM, B transposed in VGPRs.
template <typename OUT_T>
__global__ __launch_bounds__(128)
void gemm_bf16_wmma(const __bf16* __restrict__ A, const __bf16* __restrict__ Bw,
                    OUT_T* __restrict__ C, int M, int N, int K) {
    __shared__ __bf16 Alds[128 * 32];
    __shared__ __bf16 BT[64 * 32];      // [n][k] k-contiguous

    const int tid  = threadIdx.x;
    const int lane = tid & 31;
    const int w    = tid >> 5;
    const int lmod = lane & 15;
    const int kb   = (lane < 16) ? 0 : 8;
    const int lhi  = (lane < 16) ? 0 : 8;
    const int m0   = blockIdx.y * 128;
    const int n0   = blockIdx.x * 64;

    const v8f vzero = {0.f,0.f,0.f,0.f,0.f,0.f,0.f,0.f};
    v8f acc[2][4];
    #pragma unroll
    for (int ms = 0; ms < 2; ms++)
        #pragma unroll
        for (int ns = 0; ns < 4; ns++) acc[ms][ns] = vzero;

    for (int k0 = 0; k0 < K; k0 += 32) {
        // --- stage A 128x32 ---
#if HAVE_TDM
        if (tid < 32) {
            tdm_load_2d((unsigned int)(uintptr_t)Alds, &A[(size_t)m0 * K + k0],
                        (unsigned int)(K - k0), (unsigned int)(M - m0),
                        32u, 128u, (unsigned long long)K);
        }
#else
        #pragma unroll
        for (int c = tid; c < 512; c += 128) {      // 128 rows * 4 chunks
            int row = c >> 2, q = c & 3;
            *reinterpret_cast<uint4*>(&Alds[row * 32 + q * 8]) =
                *reinterpret_cast<const uint4*>(&A[(size_t)(m0 + row) * K + k0 + q * 8]);
        }
#endif
        // --- stage B 32x64 transposed into BT[n][k] ---
        #pragma unroll
        for (int c = tid; c < 256; c += 128) {
            int k = c >> 3, nc = c & 7;
            Chunk ch;
            ch.u = *reinterpret_cast<const uint4*>(&Bw[(size_t)(k0 + k) * N + n0 + nc * 8]);
            #pragma unroll
            for (int j = 0; j < 8; j++) BT[(nc * 8 + j) * 32 + k] = ch.h[j];
        }
        if (k0 + 32 < K) {  // gfx1250 global_prefetch_b8 path
            __builtin_prefetch(&Bw[(size_t)(k0 + 32 + (tid >> 2)) * N + n0], 0, 1);
        }
#if HAVE_TDM
        if (tid < 32) tdm_wait();
#endif
        __syncthreads();

        v16bf af0 = ldsFrag(&Alds[(w * 32 + lmod) * 32], kb);
        v16bf af1 = ldsFrag(&Alds[(w * 32 + 16 + lmod) * 32], kb);
        #pragma unroll
        for (int ns = 0; ns < 4; ns++) {
            v16bf bf = ldsFrag(&BT[(ns * 16 + lmod) * 32], kb);
            acc[0][ns] = wmma_bf16(af0, bf, acc[0][ns]);
            acc[1][ns] = wmma_bf16(af1, bf, acc[1][ns]);
        }
        __syncthreads();
    }

    #pragma unroll
    for (int ms = 0; ms < 2; ms++)
        #pragma unroll
        for (int ns = 0; ns < 4; ns++) {
            #pragma unroll
            for (int r = 0; r < 8; r++) {
                int row = m0 + w * 32 + ms * 16 + r + lhi;
                int col = n0 + ns * 16 + lmod;
                C[(size_t)row * N + col] = (OUT_T)acc[ms][ns][r];
            }
        }
}

// ---------------------------------------------------------------- RMSNorm + RoPE
__global__ __launch_bounds__(256)
void rmsnorm_rope(__bf16* __restrict__ buf, int rowStride,
                  const float* __restrict__ w,
                  const float* __restrict__ cosT, const float* __restrict__ sinT) {
    __shared__ float red[256];
    __shared__ float xs[256];
    const int tok  = blockIdx.x;
    const int head = blockIdx.y;
    const int d    = threadIdx.x;
    const int l    = tok & (SEQ - 1);

    __bf16* p = buf + (size_t)tok * rowStride + head * HEAD_DIM;
    float x = (float)p[d];
    red[d] = x * x;
    __syncthreads();
    #pragma unroll
    for (int s = 128; s > 0; s >>= 1) {
        if (d < s) red[d] += red[d + s];
        __syncthreads();
    }
    float inv = rsqrtf(red[0] * (1.0f / HEAD_DIM) + 1e-6f);
    float xn  = x * inv * w[d];
    xs[d] = xn;
    __syncthreads();
    float rot = (d < 128) ? -xs[d + 128] : xs[d - 128];
    float o = xn * cosT[(size_t)l * HEAD_DIM + d] + rot * sinT[(size_t)l * HEAD_DIM + d];
    p[d] = (__bf16)o;
}

// ---------------------------------------------------------------- flash attention + gate
__global__ __launch_bounds__(128)
void flash_attn_gated(const __bf16* __restrict__ QG,
                      const __bf16* __restrict__ Kb,
                      const __bf16* __restrict__ Vb,
                      __bf16* __restrict__ Og) {
    __shared__ __bf16 Qs[32 * 256];
    __shared__ __bf16 Ks[32 * 256];
    __shared__ __bf16 Vt[256 * 32];     // [d][key]
    __shared__ float  Ss[32 * 32];
    __shared__ __bf16 Ps[32 * 32];
    __shared__ float  mRow[32], lRow[32], rsRow[32];

    const int tid  = threadIdx.x;
    const int lane = tid & 31;
    const int w    = tid >> 5;
    const int lmod = lane & 15;
    const int kb   = (lane < 16) ? 0 : 8;
    const int lhi  = (lane < 16) ? 0 : 8;
    const int qs   = w & 1;    // q strip
    const int grp  = w >> 1;   // key strip for S, d-half for O
    const int h    = blockIdx.y;
    const int b    = blockIdx.z;
    const int q0   = blockIdx.x * 32;
    const int tokBase = b * SEQ;
    const int hkv  = h >> 2;   // GQA = 4

    // stage Q tile 32x256 (rows stride 2*INNER)
#if HAVE_TDM
    if (tid < 32) {
        tdm_load_2d((unsigned int)(uintptr_t)Qs,
                    &QG[(size_t)(tokBase + q0) * (2 * INNER) + h * HEAD_DIM],
                    256u, (unsigned int)(TOKENS - (tokBase + q0)),
                    256u, 32u, (unsigned long long)(2 * INNER));
        tdm_wait();
    }
#else
    #pragma unroll
    for (int c = tid; c < 1024; c += 128) {
        int row = c >> 5, cp = c & 31;
        *reinterpret_cast<uint4*>(&Qs[row * 256 + cp * 8]) =
            *reinterpret_cast<const uint4*>(
                &QG[(size_t)(tokBase + q0 + row) * (2 * INNER) + h * HEAD_DIM + cp * 8]);
    }
#endif
    if (tid < 32) { mRow[tid] = -3.0e38f; lRow[tid] = 0.f; }
    __syncthreads();

    const v8f vzero = {0.f,0.f,0.f,0.f,0.f,0.f,0.f,0.f};
    v8f oacc[8];
    #pragma unroll
    for (int t = 0; t < 8; t++) oacc[t] = vzero;

    const int nkb = q0 / 32 + 1;   // causal
    for (int kblk = 0; kblk < nkb; ++kblk) {
        const int kBase = kblk * 32;
        // stage K tile 32x256
#if HAVE_TDM
        if (tid < 32) {
            tdm_load_2d((unsigned int)(uintptr_t)Ks,
                        &Kb[(size_t)(tokBase + kBase) * KVDIM + hkv * HEAD_DIM],
                        256u, (unsigned int)(TOKENS - (tokBase + kBase)),
                        256u, 32u, (unsigned long long)KVDIM);
        }
#else
        #pragma unroll
        for (int c = tid; c < 1024; c += 128) {
            int row = c >> 5, cp = c & 31;
            *reinterpret_cast<uint4*>(&Ks[row * 256 + cp * 8]) =
                *reinterpret_cast<const uint4*>(
                    &Kb[(size_t)(tokBase + kBase + row) * KVDIM + hkv * HEAD_DIM + cp * 8]);
        }
#endif
        // stage V tile transposed 256x32 (VGPR transpose, TDM cannot transpose)
        #pragma unroll
        for (int c = tid; c < 1024; c += 128) {
            int row = c >> 5, cp = c & 31;
            Chunk ch;
            ch.u = *reinterpret_cast<const uint4*>(
                &Vb[(size_t)(tokBase + kBase + row) * KVDIM + hkv * HEAD_DIM + cp * 8]);
            #pragma unroll
            for (int j = 0; j < 8; j++) Vt[(cp * 8 + j) * 32 + row] = ch.h[j];
        }
#if HAVE_TDM
        if (tid < 32) tdm_wait();
#endif
        __syncthreads();

        // S subtile (16x16 per wave) over D=256
        v8f sacc = vzero;
        #pragma unroll
        for (int kk = 0; kk < 256; kk += 32) {
            v16bf aq = ldsFrag(&Qs[(qs * 16 + lmod) * 256 + kk], kb);
            v16bf bk = ldsFrag(&Ks[(grp * 16 + lmod) * 256 + kk], kb);
            sacc = wmma_bf16(aq, bk, sacc);
        }
        #pragma unroll
        for (int r = 0; r < 8; r++)
            Ss[(qs * 16 + r + lhi) * 32 + grp * 16 + lmod] = sacc[r];
        __syncthreads();

        // online softmax, one thread per q row
        if (tid < 32) {
            const int r  = tid;
            const int qi = q0 + r;
            float m = mRow[r];
            float rowmax = m;
            for (int c = 0; c < 32; c++) {
                float s = (kBase + c <= qi) ? Ss[r * 32 + c] * 0.0625f : -3.0e38f;
                rowmax = fmaxf(rowmax, s);
            }
            float rs = __expf(m - rowmax);
            float lacc = lRow[r] * rs;
            for (int c = 0; c < 32; c++) {
                float s = (kBase + c <= qi) ? Ss[r * 32 + c] * 0.0625f : -3.0e38f;
                float p = __expf(s - rowmax);
                Ps[r * 32 + c] = (__bf16)p;
                lacc += p;
            }
            mRow[r] = rowmax; lRow[r] = lacc; rsRow[r] = rs;
        }
        __syncthreads();

        // rescale O and accumulate P*V (wave: 16 q rows x 128 d cols)
        #pragma unroll
        for (int t = 0; t < 8; t++) {
            #pragma unroll
            for (int r = 0; r < 8; r++)
                oacc[t][r] *= rsRow[qs * 16 + r + lhi];
        }
        v16bf ap = ldsFrag(&Ps[(qs * 16 + lmod) * 32], kb);
        #pragma unroll
        for (int t = 0; t < 8; t++) {
            v16bf bv = ldsFrag(&Vt[(grp * 128 + t * 16 + lmod) * 32], kb);
            oacc[t] = wmma_bf16(ap, bv, oacc[t]);
        }
        __syncthreads();
    }

    // epilogue: 1/l normalization, silu(gate), store bf16
    #pragma unroll
    for (int t = 0; t < 8; t++) {
        #pragma unroll
        for (int r = 0; r < 8; r++) {
            int M    = qs * 16 + r + lhi;
            int row  = tokBase + q0 + M;
            int dcol = grp * 128 + t * 16 + lmod;
            float o  = oacc[t][r] / lRow[M];
            float g  = (float)QG[(size_t)row * (2 * INNER) + INNER + h * HEAD_DIM + dcol];
            float sg = g / (1.f + __expf(-g));
            Og[(size_t)row * INNER + h * HEAD_DIM + dcol] = (__bf16)(o * sg);
        }
    }
}

// ---------------------------------------------------------------- host
extern "C" void kernel_launch(void* const* d_in, const int* in_sizes, int n_in,
                              void* d_out, int out_size, void* d_ws, size_t ws_size,
                              hipStream_t stream) {
    const float* hidden = (const float*)d_in[0];
    const float* cosT   = (const float*)d_in[1];
    const float* sinT   = (const float*)d_in[2];
    const float* Wq     = (const float*)d_in[3];
    const float* Wk     = (const float*)d_in[4];
    const float* Wv     = (const float*)d_in[5];
    const float* Wo     = (const float*)d_in[6];
    const float* qnw    = (const float*)d_in[7];
    const float* knw    = (const float*)d_in[8];
    float* out = (float*)d_out;

    char* ws = (char*)d_ws;
    size_t off = 0;
    auto alloc = [&](size_t elems) -> __bf16* {
        off = (off + 255) & ~(size_t)255;
        __bf16* p = (__bf16*)(ws + off);
        off += elems * sizeof(__bf16);
        return p;
    };

    __bf16* Xbf   = alloc((size_t)TOKENS * HIDDEN);
    __bf16* Wqbf  = alloc((size_t)HIDDEN * 2 * INNER);
    __bf16* Wkbf  = alloc((size_t)HIDDEN * KVDIM);
    __bf16* Wvbf  = alloc((size_t)HIDDEN * KVDIM);
    __bf16* Wobf  = alloc((size_t)INNER * HIDDEN);
    __bf16* QG    = alloc((size_t)TOKENS * 2 * INNER);
    __bf16* Kbuf  = alloc((size_t)TOKENS * KVDIM);
    __bf16* Vbuf  = alloc((size_t)TOKENS * KVDIM);
    __bf16* AttnG = alloc((size_t)TOKENS * INNER);

    auto cvt = [&](const float* src, __bf16* dst, size_t n) {
        int blocks = (int)((n / 4 + 255) / 256);
        cvt_f32_bf16<<<blocks, 256, 0, stream>>>(src, dst, (int)n);
    };
    cvt(hidden, Xbf,  (size_t)TOKENS * HIDDEN);
    cvt(Wq,     Wqbf, (size_t)HIDDEN * 2 * INNER);
    cvt(Wk,     Wkbf, (size_t)HIDDEN * KVDIM);
    cvt(Wv,     Wvbf, (size_t)HIDDEN * KVDIM);
    cvt(Wo,     Wobf, (size_t)INNER * HIDDEN);

    // projections
    gemm_bf16_wmma<__bf16><<<dim3(2 * INNER / 64, TOKENS / 128), 128, 0, stream>>>(
        Xbf, Wqbf, QG, TOKENS, 2 * INNER, HIDDEN);
    gemm_bf16_wmma<__bf16><<<dim3(KVDIM / 64, TOKENS / 128), 128, 0, stream>>>(
        Xbf, Wkbf, Kbuf, TOKENS, KVDIM, HIDDEN);
    gemm_bf16_wmma<__bf16><<<dim3(KVDIM / 64, TOKENS / 128), 128, 0, stream>>>(
        Xbf, Wvbf, Vbuf, TOKENS, KVDIM, HIDDEN);

    // RMSNorm + RoPE (in place on q slice of QG and on Kbuf)
    rmsnorm_rope<<<dim3(TOKENS, N_HEADS), 256, 0, stream>>>(QG, 2 * INNER, qnw, cosT, sinT);
    rmsnorm_rope<<<dim3(TOKENS, N_KV), 256, 0, stream>>>(Kbuf, KVDIM, knw, cosT, sinT);

    // causal flash attention + silu(gate)
    flash_attn_gated<<<dim3(SEQ / 32, N_HEADS, BATCH), 128, 0, stream>>>(QG, Kbuf, Vbuf, AttnG);

    // output projection (fp32 result)
    gemm_bf16_wmma<float><<<dim3(HIDDEN / 64, TOKENS / 128), 128, 0, stream>>>(
        AttnG, Wobf, out, TOKENS, HIDDEN, INNER);
}
